// DrugEncoder_35699768164896
// MI455X (gfx1250) — matrix operand order repaired and checked by reading, not compiled, not run on hardware
//
#include <hip/hip_runtime.h>

// ---------------------------------------------------------------------------
// Graph TransformerConv x2 + mean pool for MI455X (gfx1250, wave32).
// GEMMs via v_wmma_f32_16x16x32_bf16 (fp32 accum), edge softmax/aggregation
// via wave-per-edge float4 gathers + L2 atomics. All scratch in d_ws.
// ---------------------------------------------------------------------------

#define N_NODES   50000
#define N_EDGES   800000
#define N_GRAPHS  512
#define D_IN      64
#define D_OUT     128   // both layers emit 128 features

typedef __attribute__((ext_vector_type(16))) __bf16        v16bf;
typedef __attribute__((ext_vector_type(8)))  float         v8f;
typedef __attribute__((ext_vector_type(4)))  unsigned int  u32x4;

union FragAB { v16bf v; u32x4 q[2]; };
union FragC  { v8f   v; float f[8]; };

// ---- helpers ---------------------------------------------------------------

__device__ __forceinline__ unsigned short f32_to_bf16(float f) {
    unsigned u = __float_as_uint(f);
    u += 0x7FFFu + ((u >> 16) & 1u);          // round to nearest even
    return (unsigned short)(u >> 16);
}

// order-preserving float -> uint key (for atomicMax on floats incl. negatives)
__device__ __forceinline__ unsigned fkey(float f) {
    unsigned u = __float_as_uint(f);
    return (u & 0x80000000u) ? ~u : (u | 0x80000000u);
}
__device__ __forceinline__ float fdecode(unsigned k) {
    unsigned u = (k & 0x80000000u) ? (k & 0x7FFFFFFFu) : ~k;
    return __uint_as_float(u);
}
#define FKEY_NEG_INF 0x007FFFFFu  // fkey(-inf)

// ---- precision conversion kernels -----------------------------------------

__global__ void convert_bf16_kernel(const float* __restrict__ in,
                                    unsigned short* __restrict__ out, int n) {
    int i = blockIdx.x * blockDim.x + threadIdx.x;
    if (i < n) out[i] = f32_to_bf16(in[i]);
}

__global__ void relu_convert_bf16_kernel(const float* __restrict__ in,
                                         unsigned short* __restrict__ out, int n) {
    int i = blockIdx.x * blockDim.x + threadIdx.x;
    if (i < n) { float v = in[i]; out[i] = f32_to_bf16(v > 0.0f ? v : 0.0f); }
}

// W[K][M] fp32 -> WT[M][K] bf16 (tiny matrices, once per launch)
__global__ void transpose_w_bf16_kernel(const float* __restrict__ W,
                                        unsigned short* __restrict__ WT,
                                        int K, int M) {
    int i = blockIdx.x * blockDim.x + threadIdx.x;
    if (i >= K * M) return;
    int k = i / M, m = i % M;
    WT[(size_t)m * K + k] = f32_to_bf16(W[i]);
}

// ---- WMMA GEMM: Out[N x 128] = X[N x K](bf16) * W + bias  (W as WT[M][K]) --

__global__ void gemm_bf16_wmma_kernel(const unsigned short* __restrict__ X,
                                      const unsigned short* __restrict__ WT,
                                      const float* __restrict__ bias,
                                      float* __restrict__ Out,
                                      int Kdim, int rowTiles, int colTiles,
                                      int Ncols) {
    const int lane = threadIdx.x & 31;
    const int wave = threadIdx.x >> 5;
    const int wavesPerBlock = blockDim.x >> 5;
    const int tile = blockIdx.x * wavesPerBlock + wave;   // uniform per wave
    if (tile >= rowTiles * colTiles) return;

    const int rowTile = tile / colTiles;
    const int colTile = tile % colTiles;
    const int mn = lane & 15;   // row (A) / col (B,C) within tile
    const int h  = lane >> 4;   // half-wave select

    const unsigned short* xrow = X  + (size_t)(rowTile * 16 + mn) * Kdim;
    const unsigned short* wrow = WT + (size_t)(colTile * 16 + mn) * Kdim;

    v8f acc = {};
    for (int kk = 0; kk < Kdim; kk += 32) {
        FragAB a, b;
        // A 16x32 bf16: elements 0..7 -> K = kk+8h.. ; 8..15 -> K = kk+16+8h..
        a.q[0] = *(const u32x4*)(xrow + kk + 8 * h);
        a.q[1] = *(const u32x4*)(xrow + kk + 16 + 8 * h);
        // B 32x16 bf16: elements 0..7 -> K = kk+8h.. ; 8..15 -> K = kk+16+8h.. (from WT row)
        b.q[0] = *(const u32x4*)(wrow + kk + 8 * h);
        b.q[1] = *(const u32x4*)(wrow + kk + 16 + 8 * h);
        acc = __builtin_amdgcn_wmma_f32_16x16x32_bf16(
            false, a.v, false, b.v, (short)0, acc, false, false);
    }

    const int col = colTile * 16 + mn;
    const float bn = bias[col];
    FragC c; c.v = acc;
    const int rbase = rowTile * 16 + 8 * h;   // C: VGPR r -> row rbase+r
    for (int r = 0; r < 8; ++r)
        Out[(size_t)(rbase + r) * Ncols + col] = c.f[r] + bn;
}

// ---- segment softmax over edges -------------------------------------------

__global__ void init_softmax_kernel(unsigned* __restrict__ mkey,
                                    float* __restrict__ ssum, int n) {
    int i = blockIdx.x * blockDim.x + threadIdx.x;
    if (i < n) { mkey[i] = FKEY_NEG_INF; ssum[i] = 0.0f; }
}

// pass A: score[e] = q[dst].k[src] / sqrt(128); atomicMax per-dst (one wave/edge)
__global__ void edge_scores_kernel(const int* __restrict__ src,
                                   const int* __restrict__ dst,
                                   const float* __restrict__ q,
                                   const float* __restrict__ k,
                                   float* __restrict__ score,
                                   unsigned* __restrict__ mkey, int E) {
    const int lane = threadIdx.x & 31;
    const int e = blockIdx.x * (blockDim.x >> 5) + (threadIdx.x >> 5);
    if (e >= E) return;
    const int s = src[e], d = dst[e];
    const float4 qv = *(const float4*)(q + (size_t)d * D_OUT + lane * 4);
    const float4 kv = *(const float4*)(k + (size_t)s * D_OUT + lane * 4);
    float sum = qv.x * kv.x + qv.y * kv.y + qv.z * kv.z + qv.w * kv.w;
    sum += __shfl_xor(sum, 16, 32);
    sum += __shfl_xor(sum, 8, 32);
    sum += __shfl_xor(sum, 4, 32);
    sum += __shfl_xor(sum, 2, 32);
    sum += __shfl_xor(sum, 1, 32);
    if (lane == 0) {
        const float sc = sum * 0.08838834764831845f;   // 1/sqrt(128)
        score[e] = sc;
        atomicMax(mkey + d, fkey(sc));
    }
}

// pass B: e[e] = exp(score - m[dst]); ssum[dst] += e  (score overwritten)
__global__ void edge_exp_kernel(const int* __restrict__ dst,
                                const unsigned* __restrict__ mkey,
                                float* __restrict__ sc_e,
                                float* __restrict__ ssum, int E) {
    int e = blockIdx.x * blockDim.x + threadIdx.x;
    if (e >= E) return;
    const int d = dst[e];
    const float m = fdecode(mkey[d]);
    const float ex = __expf(sc_e[e] - m);
    sc_e[e] = ex;
    atomicAdd(ssum + d, ex);
}

// pass C: out[dst] += (e/(ssum+eps)) * v[src]  (one wave/edge, out pre-filled
// with the skip-connection x@Ws+bs, so the += fuses the residual for free)
__global__ void edge_aggr_kernel(const int* __restrict__ src,
                                 const int* __restrict__ dst,
                                 const float* __restrict__ ew,
                                 const float* __restrict__ ssum,
                                 const float* __restrict__ v,
                                 float* __restrict__ out, int E) {
    const int lane = threadIdx.x & 31;
    const int e = blockIdx.x * (blockDim.x >> 5) + (threadIdx.x >> 5);
    if (e >= E) return;
    const int s = src[e], d = dst[e];
    float alpha = 0.0f;
    if (lane == 0) alpha = ew[e] / (ssum[d] + 1e-16f);
    alpha = __shfl(alpha, 0, 32);
    const float4 vv = *(const float4*)(v + (size_t)s * D_OUT + lane * 4);
    float* o = out + (size_t)d * D_OUT + lane * 4;
    atomicAdd(o + 0, alpha * vv.x);
    atomicAdd(o + 1, alpha * vv.y);
    atomicAdd(o + 2, alpha * vv.z);
    atomicAdd(o + 3, alpha * vv.w);
}

// ---- global mean pool ------------------------------------------------------

__global__ void pool_init_kernel(float* __restrict__ pool, float* __restrict__ cnt) {
    int i = blockIdx.x * blockDim.x + threadIdx.x;
    if (i < N_GRAPHS * D_OUT) pool[i] = 0.0f;
    if (i < N_GRAPHS) cnt[i] = 0.0f;
}

__global__ void pool_accum_kernel(const float* __restrict__ h,
                                  const int* __restrict__ batch,
                                  float* __restrict__ pool,
                                  float* __restrict__ cnt, int n) {
    const int lane = threadIdx.x & 31;
    const int node = blockIdx.x * (blockDim.x >> 5) + (threadIdx.x >> 5);
    if (node >= n) return;
    const int g = batch[node];
    const float4 hv = *(const float4*)(h + (size_t)node * D_OUT + lane * 4);
    float* p = pool + (size_t)g * D_OUT + lane * 4;
    atomicAdd(p + 0, hv.x);
    atomicAdd(p + 1, hv.y);
    atomicAdd(p + 2, hv.z);
    atomicAdd(p + 3, hv.w);
    if (lane == 0) atomicAdd(cnt + g, 1.0f);
}

__global__ void pool_final_kernel(const float* __restrict__ pool,
                                  const float* __restrict__ cnt,
                                  float* __restrict__ out) {
    int i = blockIdx.x * blockDim.x + threadIdx.x;   // N_GRAPHS*D_OUT
    if (i >= N_GRAPHS * D_OUT) return;
    out[i] = pool[i] / fmaxf(cnt[i >> 7], 1.0f);
}

// ---------------------------------------------------------------------------

extern "C" void kernel_launch(void* const* d_in, const int* in_sizes, int n_in,
                              void* d_out, int out_size, void* d_ws, size_t ws_size,
                              hipStream_t stream) {
    (void)in_sizes; (void)n_in; (void)out_size; (void)ws_size;

    const float* x     = (const float*)d_in[0];
    const int*   ei    = (const int*)  d_in[1];
    const int*   batch = (const int*)  d_in[2];
    const float* Wl[8] = { (const float*)d_in[3],  (const float*)d_in[5],
                           (const float*)d_in[7],  (const float*)d_in[9],
                           (const float*)d_in[11], (const float*)d_in[13],
                           (const float*)d_in[15], (const float*)d_in[17] };
    const float* bl[8] = { (const float*)d_in[4],  (const float*)d_in[6],
                           (const float*)d_in[8],  (const float*)d_in[10],
                           (const float*)d_in[12], (const float*)d_in[14],
                           (const float*)d_in[16], (const float*)d_in[18] };
    const int* src = ei;
    const int* dst = ei + N_EDGES;

    // ---- workspace carve-up (all re-written each call) ----
    char* ws = (char*)d_ws;
    size_t off = 0;
    auto alloc = [&](size_t bytes) -> char* {
        char* p = ws + off;
        off = (off + bytes + 255) & ~(size_t)255;
        return p;
    };
    unsigned short* wt[8];
    for (int i = 0; i < 8; ++i) {
        const int K = (i < 4) ? D_IN : D_OUT;
        wt[i] = (unsigned short*)alloc((size_t)K * D_OUT * sizeof(unsigned short));
    }
    unsigned short* xbf  = (unsigned short*)alloc((size_t)N_NODES * D_IN  * 2);
    unsigned short* hbf  = (unsigned short*)alloc((size_t)N_NODES * D_OUT * 2);
    float* qb   = (float*)alloc((size_t)N_NODES * D_OUT * 4);
    float* kb   = (float*)alloc((size_t)N_NODES * D_OUT * 4);
    float* vb   = (float*)alloc((size_t)N_NODES * D_OUT * 4);
    float* hout = (float*)alloc((size_t)N_NODES * D_OUT * 4);
    float* escr = (float*)alloc((size_t)N_EDGES * 4);
    unsigned* mkey = (unsigned*)alloc((size_t)N_NODES * 4);
    float* ssum = (float*)alloc((size_t)N_NODES * 4);
    float* pool = (float*)alloc((size_t)N_GRAPHS * D_OUT * 4);
    float* cnt  = (float*)alloc((size_t)N_GRAPHS * 4);

    // ---- weights -> bf16 transposed ----
    for (int i = 0; i < 8; ++i) {
        const int K = (i < 4) ? D_IN : D_OUT;
        const int n = K * D_OUT;
        transpose_w_bf16_kernel<<<(n + 255) / 256, 256, 0, stream>>>(Wl[i], wt[i], K, D_OUT);
    }
    convert_bf16_kernel<<<(N_NODES * D_IN) / 256, 256, 0, stream>>>(x, xbf, N_NODES * D_IN);

    const int rowTiles = N_NODES / 16;       // 3125 (exact)
    const int colTiles = D_OUT / 16;         // 8
    const int gemmBlocks = rowTiles * colTiles / 4;   // 4 waves/block (128 thr)

    const int edgeWaveBlocks = N_EDGES / 8;  // 8 waves/block (256 thr)
    const int edgeThrBlocks  = N_EDGES / 256;

    for (int layer = 0; layer < 2; ++layer) {
        const unsigned short* X = layer ? hbf : xbf;
        const int Kdim = layer ? D_OUT : D_IN;
        const int wb = layer * 4;
        // q, k, v + skip (skip GEMM writes directly into the aggregation output)
        gemm_bf16_wmma_kernel<<<gemmBlocks, 128, 0, stream>>>(X, wt[wb + 0], bl[wb + 0], qb,   Kdim, rowTiles, colTiles, D_OUT);
        gemm_bf16_wmma_kernel<<<gemmBlocks, 128, 0, stream>>>(X, wt[wb + 1], bl[wb + 1], kb,   Kdim, rowTiles, colTiles, D_OUT);
        gemm_bf16_wmma_kernel<<<gemmBlocks, 128, 0, stream>>>(X, wt[wb + 2], bl[wb + 2], vb,   Kdim, rowTiles, colTiles, D_OUT);
        gemm_bf16_wmma_kernel<<<gemmBlocks, 128, 0, stream>>>(X, wt[wb + 3], bl[wb + 3], hout, Kdim, rowTiles, colTiles, D_OUT);

        init_softmax_kernel<<<(N_NODES + 255) / 256, 256, 0, stream>>>(mkey, ssum, N_NODES);
        edge_scores_kernel<<<edgeWaveBlocks, 256, 0, stream>>>(src, dst, qb, kb, escr, mkey, N_EDGES);
        edge_exp_kernel<<<edgeThrBlocks, 256, 0, stream>>>(dst, mkey, escr, ssum, N_EDGES);
        edge_aggr_kernel<<<edgeWaveBlocks, 256, 0, stream>>>(src, dst, escr, ssum, vb, hout, N_EDGES);

        if (layer == 0)   // relu then bf16 for layer-2 GEMM inputs
            relu_convert_bf16_kernel<<<(N_NODES * D_OUT) / 256, 256, 0, stream>>>(hout, hbf, N_NODES * D_OUT);
    }

    // ---- global mean pool ----
    pool_init_kernel<<<(N_GRAPHS * D_OUT + 255) / 256, 256, 0, stream>>>(pool, cnt);
    pool_accum_kernel<<<N_NODES / 8, 256, 0, stream>>>(hout, batch, pool, cnt, N_NODES);
    pool_final_kernel<<<(N_GRAPHS * D_OUT + 255) / 256, 256, 0, stream>>>(pool, cnt, (float*)d_out);
}
